// ResGATN_72756745994561
// MI455X (gfx1250) — compile-verified, matrix operand order checked by reading, not run on hardware
//
#include <hip/hip_runtime.h>
#include <hip/hip_bf16.h>

// ---------------- problem constants (from reference) ----------------
#define NN      50000
#define NE      800000
#define F_IN    128
#define HID     96
#define OUTF    64
#define HEADS   8
#define HC      12
#define NL      6
#define EV      1000
#define ED      32
#define ACT_SLOPE 0.01f
#define ATT_SLOPE 0.2f
#define GN_EPS  1e-5f

typedef __attribute__((ext_vector_type(2)))  float    v2f;
typedef __attribute__((ext_vector_type(8)))  float    v8f;
typedef __attribute__((ext_vector_type(16))) _Float16 v16h;

#if defined(__AMDGCN__) && __has_builtin(__builtin_amdgcn_wmma_f32_16x16x4_f32)
#define USE_WMMA_F32 1
#else
#define USE_WMMA_F32 0
#endif

// ---------------- WMMA GEMM: C[M,NC] = A[M,K] @ B[K,NC] (+bias) ------------
// One wave computes a 16-row x NC-col strip; 4 waves (128 thr) per block.
// fp32 path: V_WMMA_F32_16X16X4_F32 (full precision). Fallback: f16 WMMA.
template <int K, int NC>
__global__ void __launch_bounds__(128)
gemm_wmma(const float* __restrict__ A, const float* __restrict__ B,
          const float* __restrict__ bias, float* __restrict__ C, int M)
{
    const int lane  = threadIdx.x & 31;
    const int wave  = threadIdx.x >> 5;
    const int rbase = blockIdx.x * 64 + wave * 16;
    if (rbase >= M) return;                       // wave-uniform: EXEC stays full

    int r_ld = rbase + (lane & 15);
    if (r_ld > M - 1) r_ld = M - 1;               // clamp keeps all lanes active
    const int col = lane & 15;
    const int hi  = lane >> 4;                    // 0 or 1 (half-wave)

    constexpr int NT = NC / 16;
    v8f acc[NT] = {};

#if USE_WMMA_F32
    const int kh2 = hi << 1;                      // K sub-offset 0 or 2
    for (int k0 = 0; k0 < K; k0 += 4) {
        const float* ap = A + (size_t)r_ld * K + k0 + kh2;
        v2f a; a[0] = ap[0]; a[1] = ap[1];
#pragma unroll
        for (int t = 0; t < NT; ++t) {
            const int c = t * 16 + col;
            v2f b;
            b[0] = B[(size_t)(k0 + kh2)     * NC + c];
            b[1] = B[(size_t)(k0 + kh2 + 1) * NC + c];
            acc[t] = __builtin_amdgcn_wmma_f32_16x16x4_f32(
                false, a, false, b, (short)0, acc[t], false, false);
        }
    }
#else
    const int hi8 = hi << 3;                      // K sub-offset 0 or 8
    for (int k0 = 0; k0 < K; k0 += 32) {
        const float* ap = A + (size_t)r_ld * K + k0;
        v16h a;
#pragma unroll
        for (int j = 0; j < 8; ++j) {
            a[j]     = (_Float16)ap[hi8 + j];
            a[j + 8] = (_Float16)ap[16 + hi8 + j];
        }
#pragma unroll
        for (int t = 0; t < NT; ++t) {
            const int c = t * 16 + col;
            v16h b;
#pragma unroll
            for (int j = 0; j < 8; ++j) {
                b[j]     = (_Float16)B[(size_t)(k0 + hi8 + j)      * NC + c];
                b[j + 8] = (_Float16)B[(size_t)(k0 + 16 + hi8 + j) * NC + c];
            }
            acc[t] = __builtin_amdgcn_wmma_f32_16x16x32_f16(
                false, a, false, b, (short)0, acc[t], false, false);
        }
    }
#endif

    // C/D layout: VGPR r -> row rbase+r (lanes 0-15) / rbase+r+8 (lanes 16-31)
#pragma unroll
    for (int t = 0; t < NT; ++t) {
        const int c  = t * 16 + col;
        const float bv = bias ? bias[c] : 0.f;
#pragma unroll
        for (int r = 0; r < 8; ++r) {
            const int row = rbase + r + (hi << 3);
            if (row < M) C[(size_t)row * NC + c] = acc[t][r] + bv;
        }
    }
}

// ---------------- GraphNorm folded to per-feature affine -------------------
// stats: per-feature sum and sum-of-squares over nodes
__global__ void __launch_bounds__(96)
gn_stats_k(const float* __restrict__ X, float* __restrict__ sums, int n)
{
    const int c = threadIdx.x;                    // 0..95
    const int row0 = blockIdx.x * 128;
    int rend = row0 + 128; if (rend > n) rend = n;
    float s = 0.f, s2 = 0.f;
    for (int r = row0; r < rend; ++r) {
        float v = X[(size_t)r * HID + c];
        s += v; s2 += v * v;
    }
    atomicAdd(&sums[c], s);
    atomicAdd(&sums[HID + c], s2);
}

// coef: A = w*rsqrt(var+eps), B = b - A*ms*mean, var = E[x^2]-(2ms-ms^2)mean^2
__global__ void gn_coef_k(const float* __restrict__ sums,
                          const float* __restrict__ w, const float* __restrict__ b,
                          const float* __restrict__ ms, float* __restrict__ coef,
                          float inv_n)
{
    const int c = threadIdx.x;                    // 96 threads, 1 block
    float mean = sums[c] * inv_n;
    float ex2  = sums[HID + c] * inv_n;
    float m    = ms[c];
    float var  = ex2 - (2.f * m - m * m) * mean * mean;
    float A    = w[c] * rsqrtf(var + GN_EPS);
    coef[c]        = A;
    coef[HID + c]  = fmaf(-A * m, mean, b[c]);
}

// apply affine + leaky relu (slope==1 -> identity, used for gn0)
__global__ void __launch_bounds__(256)
gn_apply_k(const float* __restrict__ H, const float* __restrict__ coef,
           float* __restrict__ T, int ntot, float slope)
{
    const int i = blockIdx.x * 256 + threadIdx.x;
    if (i >= ntot) return;
    const int c = i % HID;
    float v = fmaf(coef[c], H[i], coef[HID + c]);
    T[i] = v > 0.f ? v : v * slope;
}

// ---------------- edge kernels (gather / segment softmax / scatter) --------
__device__ __forceinline__ unsigned enc_f(float f) {
    unsigned u = __float_as_uint(f);
    return (u & 0x80000000u) ? ~u : (u | 0x80000000u);   // order-preserving
}
__device__ __forceinline__ float dec_f(unsigned u) {
    return (u & 0x80000000u) ? __uint_as_float(u & 0x7fffffffu)
                             : __uint_as_float(~u);
}

// pass A: logits per (edge, head) + segment max over dst. 4 lanes/edge,
// each lane owns 24 contiguous channels = heads 2q, 2q+1 (no x-lane reduce).
__global__ void __launch_bounds__(256)
edge_logits_k(const int* __restrict__ src, const int* __restrict__ dst,
              const int* __restrict__ code,
              const float* __restrict__ xl, const float* __restrict__ xr,
              const float* __restrict__ etab, const float* __restrict__ att,
              float* __restrict__ logits, unsigned* __restrict__ segmax, int ne)
{
    const int tid = blockIdx.x * 256 + threadIdx.x;
    const int e = tid >> 2;
    if (e >= ne) return;
    const int q = tid & 3;
    const int s = src[e], d = dst[e], cd = code[e];

    const int ep = e + 64;                        // prefetch a future gather row
    if (ep < ne) __builtin_prefetch(xl + (size_t)src[ep] * HID + q * 24, 0, 0);

    const float4* pl = (const float4*)(xl   + (size_t)s  * HID + q * 24);
    const float4* pr = (const float4*)(xr   + (size_t)d  * HID + q * 24);
    const float4* pe = (const float4*)(etab + (size_t)cd * HID + q * 24);
    const float*  aq = att + q * 24;              // heads 2q,2q+1 contiguous

    float lg0 = 0.f, lg1 = 0.f;
#pragma unroll
    for (int v = 0; v < 6; ++v) {
        float4 a = pl[v], b = pr[v], c4 = pe[v];
        float av[4] = {a.x, a.y, a.z, a.w};
        float bv[4] = {b.x, b.y, b.z, b.w};
        float cv[4] = {c4.x, c4.y, c4.z, c4.w};
#pragma unroll
        for (int j = 0; j < 4; ++j) {
            const int ch = v * 4 + j;
            float sv = av[j] + bv[j] + cv[j];
            sv = sv > 0.f ? sv : ATT_SLOPE * sv;
            float wv = aq[ch] * sv;
            if (ch < HC) lg0 += wv; else lg1 += wv;
        }
    }
    const size_t lb = (size_t)e * HEADS + q * 2;
    logits[lb]     = lg0;
    logits[lb + 1] = lg1;
    atomicMax(segmax + (size_t)d * HEADS + q * 2,     enc_f(lg0));
    atomicMax(segmax + (size_t)d * HEADS + q * 2 + 1, enc_f(lg1));
}

// pass B: p = exp(logit - m[dst]); overwrite logits with p; zsum[dst] += p
__global__ void __launch_bounds__(256)
edge_soft_k(const int* __restrict__ dst, float* __restrict__ logits,
            const unsigned* __restrict__ segmax, float* __restrict__ zsum, int neh)
{
    const int i = blockIdx.x * 256 + threadIdx.x;
    if (i >= neh) return;
    const int e = i >> 3, h = i & 7;
    const int d = dst[e];
    float m = dec_f(segmax[(size_t)d * HEADS + h]);
    float p = expf(logits[i] - m);
    logits[i] = p;
    atomicAdd(zsum + (size_t)d * HEADS + h, p);
}

// pass C: msg = xl[src] * (p/zsum * ew); scatter-add into aggr[dst]
__global__ void __launch_bounds__(256)
edge_aggr_k(const int* __restrict__ src, const int* __restrict__ dst,
            const float* __restrict__ ew, const float* __restrict__ p,
            const float* __restrict__ zsum, const float* __restrict__ xl,
            float* __restrict__ aggr, int ne)
{
    const int tid = blockIdx.x * 256 + threadIdx.x;
    const int e = tid >> 2;
    if (e >= ne) return;
    const int q = tid & 3;
    const int s = src[e], d = dst[e];
    const float w = ew[e];
    const size_t pb = (size_t)e * HEADS + q * 2;
    const size_t zb = (size_t)d * HEADS + q * 2;
    const float c0 = p[pb]     / (zsum[zb]     + 1e-16f) * w;
    const float c1 = p[pb + 1] / (zsum[zb + 1] + 1e-16f) * w;

    const float4* pl = (const float4*)(xl + (size_t)s * HID + q * 24);
    float* pa = aggr + (size_t)d * HID + q * 24;
#pragma unroll
    for (int v = 0; v < 6; ++v) {
        float4 a = pl[v];
        float av[4] = {a.x, a.y, a.z, a.w};
        const float cc = (v < 3) ? c0 : c1;       // head boundary at channel 12
#pragma unroll
        for (int j = 0; j < 4; ++j)
            atomicAdd(pa + v * 4 + j, av[j] * cc);
    }
}

// residual: h += aggr + gbias
__global__ void __launch_bounds__(256)
residual_k(float* __restrict__ h, const float* __restrict__ aggr,
           const float* __restrict__ gb, int ntot)
{
    const int i = blockIdx.x * 256 + threadIdx.x;
    if (i >= ntot) return;
    h[i] += aggr[i] + gb[i % HID];
}

// ---------------- workspace layout (all chunks naturally 256B-aligned) -----
#define BYTES_NODE ((size_t)NN * HID * 4)         // 19.2 MB
#define OFF_H      ((size_t)0)
#define OFF_T      (BYTES_NODE)
#define OFF_XL     (BYTES_NODE * 2)
#define OFF_XR     (BYTES_NODE * 3)               // aliased as aggr (see order)
#define OFF_LOGITS (BYTES_NODE * 4)
#define OFF_SEGMAX (OFF_LOGITS + (size_t)NE * HEADS * 4)
#define OFF_ZSUM   (OFF_SEGMAX + (size_t)NN * HEADS * 4)
#define OFF_ETAB   (OFF_ZSUM   + (size_t)NN * HEADS * 4)
#define OFF_STATS  (OFF_ETAB   + (size_t)EV * HID * 4)
#define OFF_COEF   (OFF_STATS  + (size_t)2 * HID * 4)
#define WS_NEED    (OFF_COEF   + (size_t)2 * HID * 4)

extern "C" void kernel_launch(void* const* d_in, const int* in_sizes, int n_in,
                              void* d_out, int out_size, void* d_ws, size_t ws_size,
                              hipStream_t stream)
{
    if (ws_size < WS_NEED) return;

    const float* x     = (const float*)d_in[0];
    const int*   ei    = (const int*)  d_in[1];
    const float* ew    = (const float*)d_in[2];
    const int*   ecode = (const int*)  d_in[3];
    const float* eemb  = (const float*)d_in[4];
    const float* linW  = (const float*)d_in[5];
    const float* linb  = (const float*)d_in[6];
    const float* gn0w  = (const float*)d_in[7];
    const float* gn0b  = (const float*)d_in[8];
    const float* gn0ms = (const float*)d_in[9];
    const float* Wl    = (const float*)d_in[10];
    const float* bl    = (const float*)d_in[11];
    const float* Wr    = (const float*)d_in[12];
    const float* br    = (const float*)d_in[13];
    const float* We    = (const float*)d_in[14];
    const float* att   = (const float*)d_in[15];
    const float* gbias = (const float*)d_in[16];
    const float* gnw   = (const float*)d_in[17];
    const float* gnb   = (const float*)d_in[18];
    const float* gnms  = (const float*)d_in[19];
    const float* loW   = (const float*)d_in[20];
    const float* lob   = (const float*)d_in[21];
    float* out = (float*)d_out;

    char* ws = (char*)d_ws;
    float*    h      = (float*)   (ws + OFF_H);
    float*    t      = (float*)   (ws + OFF_T);
    float*    xl     = (float*)   (ws + OFF_XL);
    float*    xr     = (float*)   (ws + OFF_XR);
    float*    aggr   = xr;                         // alias: xr dead after pass A
    float*    logits = (float*)   (ws + OFF_LOGITS);
    unsigned* segmax = (unsigned*)(ws + OFF_SEGMAX);
    float*    zsum   = (float*)   (ws + OFF_ZSUM);
    float*    etab   = (float*)   (ws + OFF_ETAB);
    float*    stats  = (float*)   (ws + OFF_STATS);
    float*    coef   = (float*)   (ws + OFF_COEF);

    const int* srcv = ei;
    const int* dstv = ei + NE;

    const int gGemmN = (NN + 63) / 64;
    const int gGemmE = (EV + 63) / 64;
    const int gStats = (NN + 127) / 128;
    const int gNode  = (NN * HID + 255) / 256;
    const int gE4    = (NE * 4 + 255) / 256;
    const int gE8    = (NE * HEADS + 255) / 256;
    const float invN = 1.0f / (float)NN;

    // ---- input projection + GraphNorm0 ----
    gemm_wmma<F_IN, HID><<<gGemmN, 128, 0, stream>>>(x, linW, linb, t, NN);
    hipMemsetAsync(stats, 0, 2 * HID * 4, stream);
    gn_stats_k<<<gStats, 96, 0, stream>>>(t, stats, NN);
    gn_coef_k<<<1, 96, 0, stream>>>(stats, gn0w, gn0b, gn0ms, coef, invN);
    gn_apply_k<<<gNode, 256, 0, stream>>>(t, coef, h, NN * HID, 1.0f); // slope 1 = identity

    // ---- 6 residual GATv2 layers ----
    for (int l = 0; l < NL; ++l) {
        // norm + act
        hipMemsetAsync(stats, 0, 2 * HID * 4, stream);
        gn_stats_k<<<gStats, 96, 0, stream>>>(h, stats, NN);
        gn_coef_k<<<1, 96, 0, stream>>>(stats, gnw + l * HID, gnb + l * HID,
                                        gnms + l * HID, coef, invN);
        gn_apply_k<<<gNode, 256, 0, stream>>>(h, coef, t, NN * HID, ACT_SLOPE);

        // node transforms (WMMA) + edge table T = edge_emb @ We[l]  [1000,96]
        gemm_wmma<HID, HID><<<gGemmN, 128, 0, stream>>>(t, Wl + (size_t)l*HID*HID, bl + l*HID, xl, NN);
        gemm_wmma<HID, HID><<<gGemmN, 128, 0, stream>>>(t, Wr + (size_t)l*HID*HID, br + l*HID, xr, NN);
        gemm_wmma<ED,  HID><<<gGemmE, 128, 0, stream>>>(eemb, We + (size_t)l*ED*HID, nullptr, etab, EV);

        // attention: logits -> segment max -> softmax -> weighted scatter
        hipMemsetAsync(segmax, 0, (size_t)NN * HEADS * 4, stream); // enc(-inf)<any finite
        hipMemsetAsync(zsum,   0, (size_t)NN * HEADS * 4, stream);
        edge_logits_k<<<gE4, 256, 0, stream>>>(srcv, dstv, ecode, xl, xr, etab,
                                               att + l * HEADS * HC, logits, segmax, NE);
        hipMemsetAsync(aggr, 0, BYTES_NODE, stream);  // aggr==xr, dead after pass A
        edge_soft_k<<<gE8, 256, 0, stream>>>(dstv, logits, segmax, zsum, NE * HEADS);
        edge_aggr_k<<<gE4, 256, 0, stream>>>(srcv, dstv, ew, logits, zsum, xl, aggr, NE);

        residual_k<<<gNode, 256, 0, stream>>>(h, aggr, gbias + l * HID, NN * HID);
    }

    // ---- output projection ----
    gemm_wmma<HID, OUTF><<<gGemmN, 128, 0, stream>>>(h, loW, lob, out, NN);
}
